// VMambaBlock_73452530696833
// MI455X (gfx1250) — matrix operand order repaired and checked by reading, not compiled, not run on hardware
//
#include <hip/hip_runtime.h>
#include <hip/hip_bf16.h>

// VMamba block for MI455X (gfx1250, wave32, WMMA, async-to-LDS).
// Dims: B=16, C=96, H=W=64 (L=4096), D_IN=192, N=16, R=6, K=2. Tokens=65536.

typedef __bf16 bf16;
typedef __attribute__((ext_vector_type(16))) __bf16 v16bf;
typedef __attribute__((ext_vector_type(8)))  float  v8f;

#if __has_builtin(__builtin_amdgcn_global_load_async_to_lds_b128) && \
    __has_builtin(__builtin_amdgcn_s_wait_asynccnt)
#define USE_ASYNC_LDS 1
typedef int v4i __attribute__((vector_size(16)));
typedef __attribute__((address_space(1))) v4i* gv4i_p;   // global src
typedef __attribute__((address_space(3))) v4i* lv4i_p;   // LDS dst
// Generic->AS1 is value-identical for global pointers; generic LDS addresses
// carry the LDS byte offset in the low 32 bits (ISA aperture rules), so an
// integer truncation yields the correct AS3 pointer.
__device__ __forceinline__ gv4i_p to_global_v4i(const void* p) {
  return (gv4i_p)(unsigned long long)p;
}
__device__ __forceinline__ lv4i_p to_lds_v4i(const void* p) {
  return (lv4i_p)(unsigned int)(unsigned long long)p;
}
#else
#define USE_ASYNC_LDS 0
#endif

__device__ __forceinline__ float wsum32(float v) {
#pragma unroll
  for (int m = 16; m >= 1; m >>= 1) v += __shfl_xor(v, m, 32);
  return v;
}

// A fragment (16x32 bf16, row-major source), contiguous k = 16*hi + i, same order as B.
__device__ __forceinline__ v16bf load_afrag_bf16(const bf16* __restrict__ arow,
                                                 int k0, int hi) {
  v16bf a;
#pragma unroll
  for (int i = 0; i < 16; ++i) a[i] = arow[k0 + hi * 16 + i];
  return a;
}
__device__ __forceinline__ v16bf load_afrag_f32(const float* __restrict__ arow,
                                                int k0, int hi) {
  v16bf a;
#pragma unroll
  for (int i = 0; i < 16; ++i) a[i] = (bf16)arow[k0 + hi * 16 + i];
  return a;
}
// B fragment (32x16 bf16): lane holds N = lane&15, K = 16*(lane>=16) + i.
__device__ __forceinline__ v16bf load_bfrag_f32(const float* __restrict__ wc) {
  v16bf b;
#pragma unroll
  for (int i = 0; i < 16; ++i) b[i] = (bf16)wc[i];
  return b;
}

// ---------------------------------------------------------------------------
// K0: LayerNorm over C=96 per token -> bf16 token-major.
// ---------------------------------------------------------------------------
__global__ void k_ln_in(const float* __restrict__ x,
                        const float* __restrict__ ln_w,
                        const float* __restrict__ ln_b,
                        bf16* __restrict__ hln) {
  int lane  = threadIdx.x & 31;
  int wv    = threadIdx.x >> 5;
  int token = blockIdx.x * 8 + wv;
  int b = token >> 12;
  int l = token & 4095;
  const float* xb = x + (size_t)b * 96 * 4096 + l;
  float v0 = xb[(size_t)(lane     ) * 4096];
  float v1 = xb[(size_t)(lane + 32) * 4096];
  float v2 = xb[(size_t)(lane + 64) * 4096];
  float mu = wsum32(v0 + v1 + v2) * (1.0f / 96.0f);
  float d0 = v0 - mu, d1 = v1 - mu, d2 = v2 - mu;
  float var = wsum32(d0 * d0 + d1 * d1 + d2 * d2) * (1.0f / 96.0f);
  float rs = rsqrtf(var + 1e-5f);
  bf16* hb = hln + (size_t)token * 96;
  hb[lane     ] = (bf16)(d0 * rs * ln_w[lane     ] + ln_b[lane     ]);
  hb[lane + 32] = (bf16)(d1 * rs * ln_w[lane + 32] + ln_b[lane + 32]);
  hb[lane + 64] = (bf16)(d2 * rs * ln_w[lane + 64] + ln_b[lane + 64]);
}

// ---------------------------------------------------------------------------
// K1: in_proj GEMM xz[65536,384] = hln[65536,96] * W^T.  One wave computes a
// 16x64 slab: A fragment loaded once per k-step, 4 WMMAs (A reuse x4).
// ---------------------------------------------------------------------------
__global__ void k_inproj(const bf16* __restrict__ hln,
                         const float* __restrict__ Wp,    // (384,96)
                         float* __restrict__ xz) {
  int lane = threadIdx.x & 31;
  int wv   = threadIdx.x >> 5;
  int tile = blockIdx.x * 8 + wv;           // 4096 * 6 slabs
  int mt = tile / 6, ng = tile % 6;         // ng covers cols ng*64..+63
  int n  = lane & 15, hi = lane >> 4;
  v8f acc[4] = {};
  const bf16* arow = hln + (size_t)(mt * 16 + n) * 96;
#pragma unroll
  for (int k0 = 0; k0 < 96; k0 += 32) {
    v16bf a = load_afrag_bf16(arow, k0, hi);
#pragma unroll
    for (int j = 0; j < 4; ++j) {
      int ncol = ng * 64 + j * 16 + n;
      v16bf bf = load_bfrag_f32(Wp + (size_t)ncol * 96 + k0 + hi * 16);
      acc[j] = __builtin_amdgcn_wmma_f32_16x16x32_bf16(false, a, false, bf,
                                                       (short)0, acc[j], false, false);
    }
  }
#pragma unroll
  for (int j = 0; j < 4; ++j) {
    float* ob = xz + (size_t)(mt * 16 + hi * 8) * 384 + ng * 64 + j * 16 + n;
#pragma unroll
    for (int r = 0; r < 8; ++r) ob[(size_t)r * 384] = acc[j][r];
  }
}

// ---------------------------------------------------------------------------
// K2: depthwise 3x3 conv + SiLU -> u in (b,l,d).
// ---------------------------------------------------------------------------
__global__ void k_conv(const float* __restrict__ xz,
                       const float* __restrict__ cw,
                       const float* __restrict__ cb,
                       float* __restrict__ u) {
  int t = blockIdx.x * 256 + threadIdx.x;   // < 65536*192
  int d = t % 192;
  int token = t / 192;
  int b = token >> 12, l = token & 4095;
  int h = l >> 6, w = l & 63;
  float acc = cb[d];
  const float* wk = cw + d * 9;
#pragma unroll
  for (int dh = -1; dh <= 1; ++dh) {
    int hh = h + dh;
    if ((unsigned)hh >= 64u) continue;
#pragma unroll
    for (int dw = -1; dw <= 1; ++dw) {
      int w2 = w + dw;
      if ((unsigned)w2 >= 64u) continue;
      int tk = (b << 12) + (hh << 6) + w2;
      acc += wk[(dh + 1) * 3 + (dw + 1)] * xz[(size_t)tk * 384 + d];
    }
  }
  u[(size_t)token * 192 + d] = acc / (1.0f + __expf(-acc));
}

// ---------------------------------------------------------------------------
// K3: x_proj GEMM, wave computes a 16x48 slab (3 WMMAs/k-step, N padded to 48).
// Epilogue scatters columns to dtr / Bs / Cs (scan-index layout).
// ---------------------------------------------------------------------------
__global__ void k_xproj(const float* __restrict__ u,
                        const float* __restrict__ Wx,     // (2,38,192)
                        float* __restrict__ dtr,
                        float* __restrict__ Bsb,
                        float* __restrict__ Csb) {
  int lane = threadIdx.x & 31;
  int wv   = threadIdx.x >> 5;
  int tile = blockIdx.x * 8 + wv;           // 32 * 256 slabs
  int mt = tile & 255;
  int bk = tile >> 8;                        // 0..31
  int k = bk & 1, b = bk >> 1;
  int n = lane & 15, hi = lane >> 4;
  const float* Wk = Wx + (size_t)k * 38 * 192;
  int l_row = mt * 16 + n;
  int lk = k ? (4095 - l_row) : l_row;
  const float* arow = u + (size_t)(b * 4096 + lk) * 192;
  v8f acc[3] = {};
#pragma unroll
  for (int k0 = 0; k0 < 192; k0 += 32) {
    v16bf a = load_afrag_f32(arow, k0, hi);
#pragma unroll
    for (int j = 0; j < 3; ++j) {
      int ncol = j * 16 + n;
      v16bf bf;
      if (ncol < 38) {
        bf = load_bfrag_f32(Wk + (size_t)ncol * 192 + k0 + hi * 16);
      } else {
#pragma unroll
        for (int i = 0; i < 16; ++i) bf[i] = (bf16)0.0f;
      }
      acc[j] = __builtin_amdgcn_wmma_f32_16x16x32_bf16(false, a, false, bf,
                                                       (short)0, acc[j], false, false);
    }
  }
#pragma unroll
  for (int j = 0; j < 3; ++j) {
    int ncol = j * 16 + n;
    if (ncol >= 38) continue;
#pragma unroll
    for (int r = 0; r < 8; ++r) {
      int l = mt * 16 + hi * 8 + r;
      size_t base = (size_t)bk * 4096 + l;
      float v = acc[j][r];
      if (ncol < 6)       dtr[base * 6  + ncol]        = v;
      else if (ncol < 22) Bsb[base * 16 + (ncol - 6)]  = v;
      else                Csb[base * 16 + (ncol - 22)] = v;
    }
  }
}

// ---------------------------------------------------------------------------
// K4: dt_proj (R=6) + bias + softplus -> dts (b,k,l,d).
// ---------------------------------------------------------------------------
__global__ void k_dtproj(const float* __restrict__ dtr,
                         const float* __restrict__ Wd,    // (2,192,6)
                         const float* __restrict__ bd,    // (2,192)
                         float* __restrict__ dts) {
  int t = blockIdx.x * 256 + threadIdx.x;   // 32*4096*192
  int d = t % 192;
  int rest = t / 192;
  int l = rest & 4095;
  int bk = rest >> 12;
  int k = bk & 1;
  const float* wr = Wd + (size_t)(k * 192 + d) * 6;
  const float* dr = dtr + (size_t)(bk * 4096 + l) * 6;
  float s = bd[k * 192 + d];
#pragma unroll
  for (int r = 0; r < 6; ++r) s += wr[r] * dr[r];
  float sp = (s > 20.0f) ? s : __logf(1.0f + __expf(s));
  dts[(size_t)(bk * 4096 + l) * 192 + d] = sp;
}

// ---------------------------------------------------------------------------
// K5: selective scan.  192 blocks x 32 lanes: block = (b,k,d-group of 32).
// B/C vectors double-buffered into LDS in 64-step chunks with
// GLOBAL_LOAD_ASYNC_TO_LDS_B128 (ASYNCcnt), fenced by s_wait_asynccnt 16
// so the next chunk streams while the current one is consumed.
// ---------------------------------------------------------------------------
__global__ void k_scan(const float* __restrict__ u,
                       const float* __restrict__ dts,
                       const float* __restrict__ Bsb,
                       const float* __restrict__ Csb,
                       const float* __restrict__ A_log,   // (2,192,16)
                       const float* __restrict__ Dsk,     // (2,192)
                       float* __restrict__ ys) {
  int blk  = blockIdx.x;                    // 0..191
  int bk   = blk / 6;
  int dg   = blk % 6;
  int lane = threadIdx.x;                   // 0..31
  int k = bk & 1, b = bk >> 1;
  int d = dg * 32 + lane;

  float An[16];
  const float* al = A_log + (size_t)(k * 192 + d) * 16;
#pragma unroll
  for (int n = 0; n < 16; ++n) An[n] = -__expf(al[n]);
  float Dd = Dsk[k * 192 + d];
  float st[16];
#pragma unroll
  for (int n = 0; n < 16; ++n) st[n] = 0.0f;
  size_t seq = (size_t)bk * 4096;

#if USE_ASYNC_LDS
  __shared__ alignas(16) float lbuf[2][2048];   // [buf][B:0..1023 | C:1024..2047]
  auto issue_chunk = [&](int buf, int chunk) {
    const float* gB = Bsb + (seq + (size_t)chunk * 64) * 16;
    const float* gC = Csb + (seq + (size_t)chunk * 64) * 16;
    float* lB = &lbuf[buf][0];
    float* lC = &lbuf[buf][1024];
#pragma unroll
    for (int i = 0; i < 8; ++i) {
      int off = (i * 32 + lane) * 4;        // 16B per lane per op
      __builtin_amdgcn_global_load_async_to_lds_b128(
          to_global_v4i(gB + off), to_lds_v4i(lB + off), 0, 0);
      __builtin_amdgcn_global_load_async_to_lds_b128(
          to_global_v4i(gC + off), to_lds_v4i(lC + off), 0, 0);
    }
  };
  issue_chunk(0, 0);
  int buf = 0;
  for (int chunk = 0; chunk < 64; ++chunk) {
    if (chunk + 1 < 64) {
      issue_chunk(buf ^ 1, chunk + 1);
      __builtin_amdgcn_s_wait_asynccnt(16);  // older 16 ops (this chunk) done
    } else {
      __builtin_amdgcn_s_wait_asynccnt(0);
    }
    for (int li = 0; li < 64; ++li) {
      int l = chunk * 64 + li;
      int lk = k ? (4095 - l) : l;
      float uv = u[((size_t)b * 4096 + lk) * 192 + d];
      float dt = dts[(seq + l) * 192 + d];
      if (l + 1 < 4096) {
        int lk2 = k ? (4094 - l) : (l + 1);
        __builtin_prefetch(&u[((size_t)b * 4096 + lk2) * 192 + d], 0, 0);
        __builtin_prefetch(&dts[(seq + l + 1) * 192 + d], 0, 0);
      }
      const float4* Bp = (const float4*)&lbuf[buf][li * 16];
      const float4* Cp = (const float4*)&lbuf[buf][1024 + li * 16];
      float du = dt * uv;
      float y = 0.0f;
#pragma unroll
      for (int q = 0; q < 4; ++q) {
        float4 bv = Bp[q], cv = Cp[q];
        int n0 = q * 4;
        st[n0+0] = st[n0+0] * __expf(dt * An[n0+0]) + du * bv.x; y += st[n0+0] * cv.x;
        st[n0+1] = st[n0+1] * __expf(dt * An[n0+1]) + du * bv.y; y += st[n0+1] * cv.y;
        st[n0+2] = st[n0+2] * __expf(dt * An[n0+2]) + du * bv.z; y += st[n0+2] * cv.z;
        st[n0+3] = st[n0+3] * __expf(dt * An[n0+3]) + du * bv.w; y += st[n0+3] * cv.w;
      }
      y += Dd * uv;
      ys[(seq + l) * 192 + d] = y;
    }
    buf ^= 1;
  }
#else
  for (int l = 0; l < 4096; ++l) {
    int lk = k ? (4095 - l) : l;
    float uv = u[((size_t)b * 4096 + lk) * 192 + d];
    float dt = dts[(seq + l) * 192 + d];
    if (l + 1 < 4096) {
      int lk2 = k ? (4094 - l) : (l + 1);
      __builtin_prefetch(&u[((size_t)b * 4096 + lk2) * 192 + d], 0, 0);
      __builtin_prefetch(&dts[(seq + l + 1) * 192 + d], 0, 0);
    }
    const float4* Bp = (const float4*)(Bsb + (seq + l) * 16);
    const float4* Cp = (const float4*)(Csb + (seq + l) * 16);
    float du = dt * uv;
    float y = 0.0f;
#pragma unroll
    for (int q = 0; q < 4; ++q) {
      float4 bv = Bp[q], cv = Cp[q];
      int n0 = q * 4;
      st[n0+0] = st[n0+0] * __expf(dt * An[n0+0]) + du * bv.x; y += st[n0+0] * cv.x;
      st[n0+1] = st[n0+1] * __expf(dt * An[n0+1]) + du * bv.y; y += st[n0+1] * cv.y;
      st[n0+2] = st[n0+2] * __expf(dt * An[n0+2]) + du * bv.z; y += st[n0+2] * cv.z;
      st[n0+3] = st[n0+3] * __expf(dt * An[n0+3]) + du * bv.w; y += st[n0+3] * cv.w;
    }
    y += Dd * uv;
    ys[(seq + l) * 192 + d] = y;
  }
#endif
}

// ---------------------------------------------------------------------------
// K6: cross-merge + out-LayerNorm(D=192) + SiLU gate -> bf16.
// ---------------------------------------------------------------------------
__global__ void k_merge(const float* __restrict__ ys,
                        const float* __restrict__ xz,
                        const float* __restrict__ onw,
                        const float* __restrict__ onb,
                        bf16* __restrict__ yg) {
  int lane  = threadIdx.x & 31;
  int wv    = threadIdx.x >> 5;
  int token = blockIdx.x * 8 + wv;
  int b = token >> 12, l = token & 4095;
  float v[6];
  float s = 0.0f;
#pragma unroll
  for (int j = 0; j < 6; ++j) {
    int d = lane + 32 * j;
    float y0 = ys[((size_t)(b * 2 + 0) * 4096 + l)          * 192 + d];
    float y1 = ys[((size_t)(b * 2 + 1) * 4096 + (4095 - l)) * 192 + d];
    v[j] = y0 + y1;
    s += v[j];
  }
  float mu = wsum32(s) * (1.0f / 192.0f);
  float q = 0.0f;
#pragma unroll
  for (int j = 0; j < 6; ++j) { float dd = v[j] - mu; q += dd * dd; }
  float rs = rsqrtf(wsum32(q) * (1.0f / 192.0f) + 1e-5f);
  const float* zrow = xz + (size_t)token * 384 + 192;
#pragma unroll
  for (int j = 0; j < 6; ++j) {
    int d = lane + 32 * j;
    float t = (v[j] - mu) * rs * onw[d] + onb[d];
    float z = zrow[d];
    float g = t * (z / (1.0f + __expf(-z)));
    yg[(size_t)token * 192 + d] = (bf16)g;
  }
}

// ---------------------------------------------------------------------------
// K7: out_proj GEMM; one wave computes full 16x96 row-slab (6 WMMAs/k-step),
// fused residual + gamma, writes NCHW f32.
// ---------------------------------------------------------------------------
__global__ void k_outproj(const bf16* __restrict__ yg,
                          const float* __restrict__ Wo,   // (96,192)
                          const float* __restrict__ x,
                          const float* __restrict__ gamma,
                          float* __restrict__ out) {
  int lane = threadIdx.x & 31;
  int wv   = threadIdx.x >> 5;
  int mt   = blockIdx.x * 8 + wv;           // 4096 row tiles
  int n = lane & 15, hi = lane >> 4;
  v8f acc[6] = {};
  const bf16* arow = yg + (size_t)(mt * 16 + n) * 192;
#pragma unroll
  for (int k0 = 0; k0 < 192; k0 += 32) {
    v16bf a = load_afrag_bf16(arow, k0, hi);
#pragma unroll
    for (int j = 0; j < 6; ++j) {
      v16bf bf = load_bfrag_f32(Wo + (size_t)(j * 16 + n) * 192 + k0 + hi * 16);
      acc[j] = __builtin_amdgcn_wmma_f32_16x16x32_bf16(false, a, false, bf,
                                                       (short)0, acc[j], false, false);
    }
  }
  int token0 = mt * 16 + hi * 8;
  int b = token0 >> 12;
  int l0 = token0 & 4095;
#pragma unroll
  for (int j = 0; j < 6; ++j) {
    int c = j * 16 + n;
    const float* xr   = x   + (size_t)(b * 96 + c) * 4096 + l0;
    float*       orow = out + (size_t)(b * 96 + c) * 4096 + l0;
    float g = gamma[c];
#pragma unroll
    for (int r = 0; r < 8; ++r) orow[r] = xr[r] + g * acc[j][r];
  }
}

// ---------------------------------------------------------------------------
extern "C" void kernel_launch(void* const* d_in, const int* in_sizes, int n_in,
                              void* d_out, int out_size, void* d_ws, size_t ws_size,
                              hipStream_t stream) {
  const float* x     = (const float*)d_in[0];
  const float* ln_w  = (const float*)d_in[1];
  const float* ln_b  = (const float*)d_in[2];
  const float* Wp    = (const float*)d_in[3];   // (384,96)
  const float* cw    = (const float*)d_in[4];   // (192,1,3,3)
  const float* cb    = (const float*)d_in[5];
  const float* Wx    = (const float*)d_in[6];   // (2,38,192)
  const float* Wd    = (const float*)d_in[7];   // (2,192,6)
  const float* bd    = (const float*)d_in[8];   // (2,192)
  const float* A_log = (const float*)d_in[9];   // (2,192,16)
  const float* Dsk   = (const float*)d_in[10];  // (2,192)
  const float* onw   = (const float*)d_in[11];
  const float* onb   = (const float*)d_in[12];
  const float* Wo    = (const float*)d_in[13];  // (96,192)
  const float* gamma = (const float*)d_in[14];
  float* out = (float*)d_out;
  (void)in_sizes; (void)n_in; (void)out_size; (void)ws_size;

  char* ws = (char*)d_ws;
  size_t off = 0;
  auto walloc = [&](size_t bytes) {
    char* p = ws + off;
    off = (off + bytes + 255) & ~(size_t)255;
    return p;
  };
  bf16*  hln = (bf16*) walloc((size_t)65536 * 96  * sizeof(bf16));
  float* xz  = (float*)walloc((size_t)65536 * 384 * sizeof(float));
  float* u   = (float*)walloc((size_t)65536 * 192 * sizeof(float));
  float* dtr = (float*)walloc((size_t)32 * 4096 * 6   * sizeof(float));
  float* Bsb = (float*)walloc((size_t)32 * 4096 * 16  * sizeof(float));
  float* Csb = (float*)walloc((size_t)32 * 4096 * 16  * sizeof(float));
  float* dts = (float*)walloc((size_t)32 * 4096 * 192 * sizeof(float));
  float* ysb = (float*)walloc((size_t)32 * 4096 * 192 * sizeof(float));
  bf16*  yg  = (bf16*) walloc((size_t)65536 * 192 * sizeof(bf16));

  k_ln_in  <<<8192,  256, 0, stream>>>(x, ln_w, ln_b, hln);
  k_inproj <<<3072,  256, 0, stream>>>(hln, Wp, xz);
  k_conv   <<<49152, 256, 0, stream>>>(xz, cw, cb, u);
  k_xproj  <<<1024,  256, 0, stream>>>(u, Wx, dtr, Bsb, Csb);
  k_dtproj <<<98304, 256, 0, stream>>>(dtr, Wd, bd, dts);
  k_scan   <<<192,   32,  0, stream>>>(u, dts, Bsb, Csb, A_log, Dsk, ysb);
  k_merge  <<<8192,  256, 0, stream>>>(ysb, xz, onw, onb, yg);
  k_outproj<<<512,   256, 0, stream>>>(yg, Wo, x, gamma, out);
}
